// Encoder_55044300865888
// MI455X (gfx1250) — matrix operand order repaired and checked by reading, not compiled, not run on hardware
//
#include <hip/hip_runtime.h>

// Problem constants (fixed by the reference)
#define N_ROWS 131072
#define D_IN   256
#define MID_F  384
#define H_OUT  512
#define B_SEG  256

#define KB1 (D_IN / 32)    // 8  k-blocks in GEMM1
#define KB2 (MID_F / 32)   // 12 k-blocks in GEMM2
#define NT1 (MID_F / 16)   // 24 n-tiles in GEMM1
#define NT2 (H_OUT / 16)   // 32 n-tiles in GEMM2

typedef __attribute__((ext_vector_type(16))) __bf16 v16bf;
typedef __attribute__((ext_vector_type(8)))  float  v8f;

struct alignas(16) U4 { unsigned int a, b, c, d; };   // 128-bit, trivially copyable
struct alignas(8)  U2 { unsigned int a, b; };         // 64-bit

union FragBF { v16bf v; unsigned short u[16]; unsigned int d[8]; U4 q[2]; };
union FragF  { v8f   v; float          f[8];  };

// fp32 -> bf16 with round-to-nearest-even
__device__ __forceinline__ unsigned short f2b(float f) {
  unsigned int u = __float_as_uint(f);
  u += 0x7FFFu + ((u >> 16) & 1u);
  return (unsigned short)(u >> 16);
}
__device__ __forceinline__ unsigned int pack2(float a, float b) {
  return (unsigned int)f2b(a) | ((unsigned int)f2b(b) << 16);
}
// mish(x) = x * tanh(softplus(x)) = x * (t^2-1)/(t^2+1), t = 1+e^x.
// One v_exp_f32 + a few VALU ops instead of exp+log1p+tanh expansions.
__device__ __forceinline__ float mish(float x) {
  if (x > 20.f) return x;               // softplus ~ x, tanh ~ 1 (also avoids t^2 overflow)
  const float t  = 1.f + __expf(x);
  const float t2 = t * t;
  return x * (t2 - 1.f) / (t2 + 1.f);
}

// A 16x32 bf16 fragment slot for an aligned *even* K pair (k, k+1):
// lanes 0-15 hold K {0..15}, lanes 16-31 hold {16..31}.
// Returns (vgpr index v, lane-half kh); both elements land in one dword.
__device__ __forceinline__ void a_slot_pair(int koff, int& v, int& kh) {
  kh = koff >> 4;
  v  = (koff & 15) >> 1;
}

// ---------------------------------------------------------------------------
// 1) Pack W1 / W2 into bf16 WMMA B-fragment order:
//    dword index = ((nt*KB + kb)*32 + lane)*8 + v,
//    holding bf16 pair (W[kk][col], W[kk+1][col]),
//    kk = kb*32 + (lane>>4)*16 + 2v, col = nt*16 + (lane&15).
// ---------------------------------------------------------------------------
__global__ void pack_w_kernel(const float* __restrict__ W1,
                              const float* __restrict__ W2,
                              unsigned int* __restrict__ W1p,
                              unsigned int* __restrict__ W2p) {
  const int n1 = NT1 * KB1 * 32 * 8;   // 49152 dwords
  const int n2 = NT2 * KB2 * 32 * 8;   // 98304 dwords
  for (int t = blockIdx.x * blockDim.x + threadIdx.x; t < n1 + n2;
       t += gridDim.x * blockDim.x) {
    if (t < n1) {
      const int v = t & 7, lane = (t >> 3) & 31, blk = t >> 8;
      const int kb = blk % KB1, nt = blk / KB1;
      const int col = nt * 16 + (lane & 15);
      const int kk  = kb * 32 + (lane >> 4) * 16 + 2 * v;
      W1p[t] = pack2(W1[(size_t)kk * MID_F + col], W1[(size_t)(kk + 1) * MID_F + col]);
    } else {
      const int t2 = t - n1;
      const int v = t2 & 7, lane = (t2 >> 3) & 31, blk = t2 >> 8;
      const int kb = blk % KB2, nt = blk / KB2;
      const int col = nt * 16 + (lane & 15);
      const int kk  = kb * 32 + (lane >> 4) * 16 + 2 * v;
      W2p[t2] = pack2(W2[(size_t)kk * H_OUT + col], W2[(size_t)(kk + 1) * H_OUT + col]);
    }
  }
}

// ---------------------------------------------------------------------------
// 2) mag = x @ Wr + br  (one wave32 per row, float4 loads, shuffle reduce)
// ---------------------------------------------------------------------------
__global__ void mag_kernel(const float* __restrict__ x,
                           const float* __restrict__ Wr,
                           const float* __restrict__ br,
                           float* __restrict__ mag) {
  const int wave = threadIdx.x >> 5;
  const int lane = threadIdx.x & 31;
  const int row  = blockIdx.x * 8 + wave;
  const float* xr = x + (size_t)row * D_IN;
  float s = 0.f;
  #pragma unroll
  for (int k4 = lane; k4 < D_IN / 4; k4 += 32) {
    const float4 xv = ((const float4*)xr)[k4];
    const float4 wv = ((const float4*)Wr)[k4];
    s += xv.x * wv.x + xv.y * wv.y + xv.z * wv.z + xv.w * wv.w;
  }
  #pragma unroll
  for (int o = 16; o > 0; o >>= 1) s += __shfl_xor(s, o, 32);
  if (lane == 0) mag[row] = s + br[0];
}

// ---------------------------------------------------------------------------
// 3) Segment offsets: ptr[b] = lower_bound(batch, b) (batch is sorted)
// ---------------------------------------------------------------------------
__global__ void ptr_kernel(const int* __restrict__ batch, int* __restrict__ ptr) {
  const int b = blockIdx.x * blockDim.x + threadIdx.x;
  if (b > B_SEG) return;
  int lo = 0, hi = N_ROWS;
  while (lo < hi) {
    int mid = (lo + hi) >> 1;
    if (batch[mid] < b) lo = mid + 1; else hi = mid;
  }
  ptr[b] = lo;
}

// ---------------------------------------------------------------------------
// 4) rank[i] = stable within-segment rank of mag[i] (replaces the lexsort)
// ---------------------------------------------------------------------------
__global__ void rank_kernel(const int* __restrict__ batch,
                            const float* __restrict__ mag,
                            const int* __restrict__ ptr,
                            int* __restrict__ rank) {
  const int i = blockIdx.x * blockDim.x + threadIdx.x;
  if (i >= N_ROWS) return;
  const int s  = batch[i];
  const int lo = ptr[s], hi = ptr[s + 1];
  const float mi = mag[i];
  int r = 0;
  for (int j = lo; j < hi; ++j) {
    float mj = mag[j];
    r += (mj < mi) || (mj == mi && j < i);
  }
  rank[i] = r;
}

// ---------------------------------------------------------------------------
// 5) out[b][h] = n[b] * Wc[h] + bc[h]  (cardinality term; inits accumulator)
// ---------------------------------------------------------------------------
__global__ void init_out_kernel(const int* __restrict__ ptr,
                                const float* __restrict__ Wc,
                                const float* __restrict__ bc,
                                float* __restrict__ out) {
  const int i = blockIdx.x * blockDim.x + threadIdx.x;
  if (i >= B_SEG * H_OUT) return;
  const int b = i / H_OUT, h = i - b * H_OUT;
  const float n = (float)(ptr[b + 1] - ptr[b]);
  out[i] = n * Wc[h] + bc[h];
}

// ---------------------------------------------------------------------------
// 6) Fused: GEMM1 (bf16 WMMA) -> LayerNorm -> Mish -> GEMM2 (bf16 WMMA)
//    -> *keys[rank] -> segment-sum via global_atomic_add_f32
//    One block = 16 rows, 128 threads = 4 waves.
//    A frags: 2x ds_load_b128 ; B frags: 2x global_load_b128 (pre-packed).
//    First k-step of every tile uses WMMA SRC2 = inline 0; bias folded into
//    the epilogue. Cold sections kept rolled to protect I-cache.
// ---------------------------------------------------------------------------
__launch_bounds__(128, 2)
__global__ void fused_kernel(const float* __restrict__ x,
                             const int* __restrict__ batch,
                             const unsigned int* __restrict__ W1p,
                             const float* __restrict__ b1,
                             const float* __restrict__ g1,
                             const float* __restrict__ beta1,
                             const unsigned int* __restrict__ W2p,
                             const float* __restrict__ b2,
                             const float* __restrict__ key_table,
                             const int* __restrict__ rank,
                             float* __restrict__ out) {
  __shared__ unsigned int xA[KB1 * 32 * 8];        // x tile, A-frag order (8 KB)
  __shared__ unsigned int hA[KB2 * 32 * 8];        // mish(h), A-frag order (12 KB)
  __shared__ float        hF[16][MID_F + 8];       // h tile f32 for LN (~25 KB)

  const int tid   = threadIdx.x;
  const int wave  = tid >> 5;
  const int lane  = tid & 31;
  const int row0  = blockIdx.x * 16;
  const int mrow  = lane & 15;        // N index for B/C/D
  const int khalf = lane >> 4;

  // ---- stage x tile: global_load_b128 -> packed ds_store_b64 ------------
  // One task = 4 consecutive K of one row -> 2 adjacent frag dwords.
  #pragma unroll 1
  for (int t = tid; t < 16 * (D_IN / 4); t += 128) {
    const int m = t >> 6;               // D_IN/4 == 64 tasks per row
    const int k = (t & 63) * 4;
    const float4 xv = *(const float4*)&x[(size_t)(row0 + m) * D_IN + k];
    const int kb = k >> 5, koff = k & 31;
    int v, kh;
    a_slot_pair(koff, v, kh);           // k aligned to 4 -> v even
    const int flane = m + 16 * kh;
    U2 wr;
    wr.a = pack2(xv.x, xv.y);
    wr.b = pack2(xv.z, xv.w);
    *(U2*)&xA[(kb * 32 + flane) * 8 + v] = wr;
  }
  __syncthreads();

  // ---- hoist all GEMM1 A fragments (read LDS once per wave) -------------
  FragBF a1[KB1];
  #pragma unroll
  for (int kb = 0; kb < KB1; ++kb) {
    const U4* p = (const U4*)&xA[(kb * 32 + lane) * 8];
    a1[kb].q[0] = p[0];
    a1[kb].q[1] = p[1];
  }

  // ---- GEMM1: h = x @ W1 (+b1 in epilogue), 6 of 24 N-tiles per wave ----
  for (int t = 0; t < 6; ++t) {
    const int nt  = wave * 6 + t;
    const int col = nt * 16 + mrow;
    FragF acc;
    {
      FragBF b;
      const U4* q = (const U4*)&W1p[((nt * KB1) * 32 + lane) * 8];
      b.q[0] = q[0];
      b.q[1] = q[1];
      v8f z = {};                        // SRC2 = inline 0
      acc.v = __builtin_amdgcn_wmma_f32_16x16x32_bf16(
          false, a1[0].v, false, b.v, (short)0, z, false, false);
    }
    #pragma unroll
    for (int kb = 1; kb < KB1; ++kb) {
      FragBF b;
      const U4* q = (const U4*)&W1p[(((nt * KB1 + kb) * 32) + lane) * 8];
      b.q[0] = q[0];
      b.q[1] = q[1];
      acc.v = __builtin_amdgcn_wmma_f32_16x16x32_bf16(
          false, a1[kb].v, false, b.v, (short)0, acc.v, false, false);
    }
    const float bias = b1[col];
    #pragma unroll
    for (int r = 0; r < 8; ++r)
      hF[r + 8 * khalf][col] = acc.f[r] + bias;
  }
  __syncthreads();

  // ---- LayerNorm + Mish (each wave: 4 rows, shuffle reductions);
  //      output packed pairwise straight into A-fragment layout -----------
  for (int rr = 0; rr < 4; ++rr) {
    const int m = wave * 4 + rr;
    float s = 0.f, s2 = 0.f;
    #pragma unroll 1
    for (int j = lane; j < MID_F / 2; j += 32) {
      const float2 hv = *(const float2*)&hF[m][2 * j];
      s  += hv.x + hv.y;
      s2 += hv.x * hv.x + hv.y * hv.y;
    }
    #pragma unroll
    for (int o = 16; o > 0; o >>= 1) {
      s  += __shfl_xor(s,  o, 32);
      s2 += __shfl_xor(s2, o, 32);
    }
    const float mu  = s * (1.0f / MID_F);
    const float var = s2 * (1.0f / MID_F) - mu * mu;
    const float inv = rsqrtf(var + 1e-5f);
    #pragma unroll 1
    for (int j = lane; j < MID_F / 2; j += 32) {
      const int k = 2 * j;
      const float2 hv = *(const float2*)&hF[m][k];
      const float2 gv = *(const float2*)&g1[k];
      const float2 bv = *(const float2*)&beta1[k];
      const float va = mish((hv.x - mu) * inv * gv.x + bv.x);
      const float vb = mish((hv.y - mu) * inv * gv.y + bv.y);
      const int kb = k >> 5, koff = k & 31;
      int v, kh;
      a_slot_pair(koff, v, kh);          // even pair -> one dword
      hA[(kb * 32 + m + 16 * kh) * 8 + v] = pack2(va, vb);
    }
  }
  __syncthreads();

  // ---- per-row rank / segment loaded once -------------------------------
  int rks[8], segs[8];
  #pragma unroll
  for (int r = 0; r < 8; ++r) {
    const int gi = row0 + r + 8 * khalf;
    rks[r]  = rank[gi];
    segs[r] = batch[gi];
  }

  // ---- hoist all GEMM2 A fragments --------------------------------------
  FragBF a2[KB2];
  #pragma unroll
  for (int kb = 0; kb < KB2; ++kb) {
    const U4* p = (const U4*)&hA[(kb * 32 + lane) * 8];
    a2[kb].q[0] = p[0];
    a2[kb].q[1] = p[1];
  }

  // ---- GEMM2: y = mish(h) @ W2 (+b2), 8 of 32 N-tiles per wave ----------
  for (int t = 0; t < 8; ++t) {
    const int nt  = wave * 8 + t;
    const int col = nt * 16 + mrow;

    // speculative prefetch of next tile's packed B slab (global_prefetch_b8)
    if (t + 1 < 8)
      __builtin_prefetch((const void*)&W2p[(((nt + 1) * KB2) * 32 + lane) * 8], 0, 1);

    FragF acc;
    {
      FragBF b;
      const U4* q = (const U4*)&W2p[((nt * KB2) * 32 + lane) * 8];
      b.q[0] = q[0];
      b.q[1] = q[1];
      v8f z = {};                        // SRC2 = inline 0
      acc.v = __builtin_amdgcn_wmma_f32_16x16x32_bf16(
          false, a2[0].v, false, b.v, (short)0, z, false, false);
    }
    #pragma unroll
    for (int kb = 1; kb < KB2; ++kb) {
      FragBF b;
      const U4* q = (const U4*)&W2p[(((nt * KB2 + kb) * 32) + lane) * 8];
      b.q[0] = q[0];
      b.q[1] = q[1];
      acc.v = __builtin_amdgcn_wmma_f32_16x16x32_bf16(
          false, a2[kb].v, false, b.v, (short)0, acc.v, false, false);
    }

    // epilogue: (+bias) * key_table[rank], segment-sum via atomic f32 add
    const float bias = b2[col];
    #pragma unroll
    for (int r = 0; r < 8; ++r) {
      const float y = (acc.f[r] + bias) * key_table[(size_t)rks[r] * H_OUT + col];
      atomicAdd(&out[(size_t)segs[r] * H_OUT + col], y);
    }
  }
}

// ---------------------------------------------------------------------------
// Launch
// ---------------------------------------------------------------------------
extern "C" void kernel_launch(void* const* d_in, const int* in_sizes, int n_in,
                              void* d_out, int out_size, void* d_ws, size_t ws_size,
                              hipStream_t stream) {
  const float* x         = (const float*)d_in[0];
  const int*   batch     = (const int*)  d_in[1];
  // d_in[2] = n_batches (scalar, == B_SEG)
  const float* Wr        = (const float*)d_in[3];
  const float* br        = (const float*)d_in[4];
  const float* W1        = (const float*)d_in[5];
  const float* b1        = (const float*)d_in[6];
  const float* g1        = (const float*)d_in[7];
  const float* beta1     = (const float*)d_in[8];
  const float* W2        = (const float*)d_in[9];
  const float* b2        = (const float*)d_in[10];
  const float* key_table = (const float*)d_in[11];
  const float* Wc        = (const float*)d_in[12];
  const float* bc        = (const float*)d_in[13];
  float* out = (float*)d_out;

  // workspace layout (bytes)
  char* ws = (char*)d_ws;
  unsigned int* W1p = (unsigned int*)(ws);                 // 196608 B
  unsigned int* W2p = (unsigned int*)(ws + 196608);        // 393216 B
  float*        mag = (float*)(ws + 589824);               // 524288 B
  int*          ptr = (int*)(ws + 1114112);                //   1028 B
  int*          rnk = (int*)(ws + 1116160);                // 524288 B

  pack_w_kernel<<<512, 256, 0, stream>>>(W1, W2, W1p, W2p);
  mag_kernel<<<N_ROWS / 8, 256, 0, stream>>>(x, Wr, br, mag);
  ptr_kernel<<<2, 256, 0, stream>>>(batch, ptr);
  rank_kernel<<<N_ROWS / 256, 256, 0, stream>>>(batch, mag, ptr, rnk);
  init_out_kernel<<<(B_SEG * H_OUT) / 256, 256, 0, stream>>>(ptr, Wc, bc, out);
  fused_kernel<<<N_ROWS / 16, 128, 0, stream>>>(
      x, batch, W1p, b1, g1, beta1, W2p, b2, key_table, rnk, out);
}
